// TopoFlowAttention_6098853560439
// MI455X (gfx1250) — compile-verified, matrix operand order checked by reading, not run on hardware
//
#include <hip/hip_runtime.h>
#include <cstdint>
#include <cstddef>

typedef _Float16 half8 __attribute__((ext_vector_type(8)));
typedef _Float16 v16h  __attribute__((ext_vector_type(16)));
typedef float    v8f   __attribute__((ext_vector_type(8)));
typedef int      i4    __attribute__((ext_vector_type(4)));

#define NB   2
#define NH   12
#define NSEQ 2048
#define HD   64
#define CDIM 768
#define C3   2304

#if defined(__has_builtin)
#  if __has_builtin(__builtin_amdgcn_global_load_async_to_lds_b128) && \
      __has_builtin(__builtin_amdgcn_s_wait_asynccnt)
#    define HAVE_ASYNC_LDS 1
#  else
#    define HAVE_ASYNC_LDS 0
#  endif
#else
#  define HAVE_ASYNC_LDS 0
#endif

#if HAVE_ASYNC_LDS
#define GA_I4(p) ((__attribute__((address_space(1))) i4*)(void*)(p))
#define LA_I4(p) ((__attribute__((address_space(3))) i4*)(void*)(p))
#endif

__device__ __forceinline__ v16h mk_frag(const _Float16* p0, const _Float16* p1) {
  half8 lo = *(const half8*)p0;
  half8 hi = *(const half8*)p1;
  v16h r;
#pragma unroll
  for (int i = 0; i < 8; ++i) { r[i] = lo[i]; r[i + 8] = hi[i]; }
  return r;
}

// ---------------- elementwise f32 -> f16 ----------------
__global__ void k_f32_to_f16(const float* __restrict__ in, _Float16* __restrict__ out, int n) {
  int i = blockIdx.x * blockDim.x + threadIdx.x;
  if (i < n) out[i] = (_Float16)in[i];
}

// ---------------- wind strength: 2x2 avg pool of magnitude ----------------
__global__ void k_wind(const float* __restrict__ u, const float* __restrict__ v,
                       float* __restrict__ ws) {
  int i = blockIdx.x * blockDim.x + threadIdx.x; // B*N
  if (i >= NB * NSEQ) return;
  int b = i / NSEQ, n = i % NSEQ;
  int ph = n / 64, pw = n % 64; // pooled grid 32 x 64
  const float* ub = u + (size_t)b * 64 * 128;
  const float* vb = v + (size_t)b * 64 * 128;
  float s = 0.f;
#pragma unroll
  for (int dy = 0; dy < 2; ++dy)
#pragma unroll
    for (int dx = 0; dx < 2; ++dx) {
      int idx = (2 * ph + dy) * 128 + (2 * pw + dx);
      float uu = ub[idx], vv = vb[idx];
      s += sqrtf(uu * uu + vv * vv + 1e-8f);
    }
  ws[i] = 0.25f * s;
}

// ---------------- bias[b,i,j] precompute (f16) ----------------
__global__ void k_bias(const float* __restrict__ elev, const float* __restrict__ ws,
                       const float* __restrict__ alpha_p, const float* __restrict__ beta_p,
                       _Float16* __restrict__ bias) {
  size_t i = (size_t)blockIdx.x * blockDim.x + threadIdx.x;
  if (i >= (size_t)NB * NSEQ * NSEQ) return;
  int b = (int)(i / ((size_t)NSEQ * NSEQ));
  size_t rem = i % ((size_t)NSEQ * NSEQ);
  int ii = (int)(rem / NSEQ), jj = (int)(rem % NSEQ);
  float ei = elev[b * NSEQ + ii], ej = elev[b * NSEQ + jj];
  float diff = (ej - ei) * (1.0f / 1000.0f);
  float braw = -alpha_p[0] * fmaxf(diff, 0.f);
  float wavg = 0.5f * (ws[b * NSEQ + ii] + ws[b * NSEQ + jj]);
  float sig = 1.f / (1.f + __expf(-(wavg - 5.0f)));
  float mod = 1.f - beta_p[0] * sig;
  float val = fminf(fmaxf(braw * mod, -10.f), 0.f);
  bias[i] = (_Float16)val;
}

// ---------------- QKV projection GEMM: 32x64 per wave, ping-pong pipelined ----------------
__global__ void __launch_bounds__(128)
k_gemm_qkv(const _Float16* __restrict__ xh, const _Float16* __restrict__ wh,
           _Float16* __restrict__ qb, _Float16* __restrict__ kb, _Float16* __restrict__ vb) {
  int lane = threadIdx.x & 31;
  int wave = threadIdx.x >> 5;
  int rowBase = (blockIdx.y * 4 + wave) * 32;
  int colBase = blockIdx.x * 64;
  int lr = lane & 15;
  int koff = (lane < 16) ? 0 : 8;
  int rbase = (lane < 16) ? 0 : 8;
  const _Float16* arow0 = xh + (size_t)(rowBase + lr) * CDIM;
  const _Float16* arow1 = xh + (size_t)(rowBase + 16 + lr) * CDIM;
  const _Float16* wrow[4];
#pragma unroll
  for (int t = 0; t < 4; ++t) wrow[t] = wh + (size_t)(colBase + t * 16 + lr) * CDIM;

  v8f acc[2][4] = {};
  v16h aF[2][2], bF[2][4];
  aF[0][0] = mk_frag(arow0 + koff, arow0 + koff + 16);
  aF[0][1] = mk_frag(arow1 + koff, arow1 + koff + 16);
#pragma unroll
  for (int t = 0; t < 4; ++t) bF[0][t] = mk_frag(wrow[t] + koff, wrow[t] + koff + 16);

  for (int k0 = 0; k0 < CDIM; k0 += 64) {
    int k1 = k0 + 32; // CDIM % 64 == 0 -> always in range
    aF[1][0] = mk_frag(arow0 + k1 + koff, arow0 + k1 + koff + 16);
    aF[1][1] = mk_frag(arow1 + k1 + koff, arow1 + k1 + koff + 16);
#pragma unroll
    for (int t = 0; t < 4; ++t) bF[1][t] = mk_frag(wrow[t] + k1 + koff, wrow[t] + k1 + koff + 16);
#pragma unroll
    for (int t = 0; t < 4; ++t) {
      acc[0][t] = __builtin_amdgcn_wmma_f32_16x16x32_f16(false, aF[0][0], false, bF[0][t],
                                                         (short)0, acc[0][t], false, false);
      acc[1][t] = __builtin_amdgcn_wmma_f32_16x16x32_f16(false, aF[0][1], false, bF[0][t],
                                                         (short)0, acc[1][t], false, false);
    }
    int k2 = (k0 + 64 < CDIM) ? (k0 + 64) : 0; // wrapped prefetch on final stage
    aF[0][0] = mk_frag(arow0 + k2 + koff, arow0 + k2 + koff + 16);
    aF[0][1] = mk_frag(arow1 + k2 + koff, arow1 + k2 + koff + 16);
#pragma unroll
    for (int t = 0; t < 4; ++t) bF[0][t] = mk_frag(wrow[t] + k2 + koff, wrow[t] + k2 + koff + 16);
#pragma unroll
    for (int t = 0; t < 4; ++t) {
      acc[0][t] = __builtin_amdgcn_wmma_f32_16x16x32_f16(false, aF[1][0], false, bF[1][t],
                                                         (short)0, acc[0][t], false, false);
      acc[1][t] = __builtin_amdgcn_wmma_f32_16x16x32_f16(false, aF[1][1], false, bF[1][t],
                                                         (short)0, acc[1][t], false, false);
    }
  }
#pragma unroll
  for (int g = 0; g < 2; ++g)
#pragma unroll
    for (int t = 0; t < 4; ++t) {
      int c = colBase + t * 16 + lr;
      int part = c / CDIM;
      int within = c % CDIM;
      int head = within / HD;
      int d = within % HD;
      _Float16* dst = (part == 0) ? qb : ((part == 1) ? kb : vb);
#pragma unroll
      for (int r = 0; r < 8; ++r) {
        int m = rowBase + g * 16 + rbase + r;
        int b = m / NSEQ;
        int n = m % NSEQ;
        dst[(((size_t)b * NH + head) * NSEQ + n) * HD + d] = (_Float16)acc[g][t][r];
      }
    }
}

// ---------------- Flash attention per (b,h): ping-pong K/V tiles ----------------
__global__ void __launch_bounds__(128)
k_attn(const _Float16* __restrict__ qb, const _Float16* __restrict__ kb,
       const _Float16* __restrict__ vb, const _Float16* __restrict__ bias,
       _Float16* __restrict__ attn_out) {
  __shared__ _Float16 Kt[2][32 * 64];      // [j][hd] row-major
  __shared__ _Float16 Vt[2][64 * 32];      // [hd][j] transposed
  __shared__ _Float16 Pst[4][32 * 16];     // per-wave, col-major [col][row]

  int lane = threadIdx.x & 31;
  int wave = threadIdx.x >> 5;
  int tid = threadIdx.x;
  int bh = blockIdx.y;
  int b = bh / NH;
  int qbase = blockIdx.x * 64 + wave * 16;
  int lr = lane & 15;
  int koff = (lane < 16) ? 0 : 8;
  int rbase = (lane < 16) ? 0 : 8;

  const _Float16* qrow = qb + ((size_t)bh * NSEQ + qbase + lr) * HD;
  v16h qa0 = mk_frag(qrow + koff, qrow + koff + 16);
  v16h qa1 = mk_frag(qrow + 32 + koff, qrow + 32 + koff + 16);

  v8f oacc[4] = {};
  float mrun[8], lrun[8];
#pragma unroll
  for (int r = 0; r < 8; ++r) { mrun[r] = -1e30f; lrun[r] = 0.f; }

  const _Float16* kg0 = kb + (size_t)bh * NSEQ * HD;
  const _Float16* vg0 = vb + (size_t)bh * NSEQ * HD;
  const float scale = 0.125f; // 64^-0.5

  int trow[2], tch[2];
#pragma unroll
  for (int it = 0; it < 2; ++it) {
    int idx = tid + it * 128;
    trow[it] = idx >> 3;
    tch[it] = (idx & 7) * 8;
  }

  // ---- prologue: fill buffer 0 with tile jt=0 ----
#pragma unroll
  for (int it = 0; it < 2; ++it) {
#if HAVE_ASYNC_LDS
    __builtin_amdgcn_global_load_async_to_lds_b128(
        GA_I4(kg0 + (size_t)trow[it] * HD + tch[it]),
        LA_I4(&Kt[0][trow[it] * 64 + tch[it]]), 0, 0);
#else
    *(half8*)&Kt[0][trow[it] * 64 + tch[it]] =
        *(const half8*)(kg0 + (size_t)trow[it] * HD + tch[it]);
#endif
    half8 vv = *(const half8*)(vg0 + (size_t)trow[it] * HD + tch[it]);
#pragma unroll
    for (int e = 0; e < 8; ++e) Vt[0][(tch[it] + e) * 32 + trow[it]] = vv[e];
  }
#if HAVE_ASYNC_LDS
  __builtin_amdgcn_s_wait_asynccnt(0);
#endif
  __syncthreads();

  auto step = [&](int jt, int cur, int nxt) {
    int jn = (jt + 32 < NSEQ) ? (jt + 32) : 0; // wrapped prefetch (last-iter write unused)

    // ---- issue next-tile fetches (overlap with compute below) ----
#if HAVE_ASYNC_LDS
#pragma unroll
    for (int it = 0; it < 2; ++it)
      __builtin_amdgcn_global_load_async_to_lds_b128(
          GA_I4(kg0 + (size_t)(jn + trow[it]) * HD + tch[it]),
          LA_I4(&Kt[nxt][trow[it] * 64 + tch[it]]), 0, 0);
    half8 kreg[2]; // unused in async path
    (void)kreg;
    half8 vreg[2];
#pragma unroll
    for (int it = 0; it < 2; ++it)
      vreg[it] = *(const half8*)(vg0 + (size_t)(jn + trow[it]) * HD + tch[it]);
#else
    half8 kreg[2], vreg[2];
#pragma unroll
    for (int it = 0; it < 2; ++it) {
      kreg[it] = *(const half8*)(kg0 + (size_t)(jn + trow[it]) * HD + tch[it]);
      vreg[it] = *(const half8*)(vg0 + (size_t)(jn + trow[it]) * HD + tch[it]);
    }
#endif

    // ---- hoist V B-fragments (ds loads overlap score WMMAs) ----
    v16h vfr[4];
#pragma unroll
    for (int t = 0; t < 4; ++t) {
      const _Float16* vcol = &Vt[cur][(t * 16 + lr) * 32];
      vfr[t] = mk_frag(vcol + koff, vcol + koff + 16);
    }

    // ---- scores for two 16-col subtiles ----
    float sc[2][8];
#pragma unroll
    for (int s = 0; s < 2; ++s) {
      const _Float16* krow = &Kt[cur][(s * 16 + lr) * 64];
      v16h b0 = mk_frag(krow + koff, krow + koff + 16);
      v16h b1 = mk_frag(krow + 32 + koff, krow + 32 + koff + 16);
      v8f a = {};
      a = __builtin_amdgcn_wmma_f32_16x16x32_f16(false, qa0, false, b0, (short)0, a, false, false);
      a = __builtin_amdgcn_wmma_f32_16x16x32_f16(false, qa1, false, b1, (short)0, a, false, false);
      int j = jt + s * 16 + lr;
      const _Float16* bp = bias + ((size_t)b * NSEQ + (qbase + rbase)) * NSEQ + j;
#pragma unroll
      for (int r = 0; r < 8; ++r) sc[s][r] = a[r] * scale + (float)bp[(size_t)r * NSEQ];
    }

    // ---- online softmax ----
    float p0h[8], p1h[8];
#pragma unroll
    for (int r = 0; r < 8; ++r) {
      float t = fmaxf(sc[0][r], sc[1][r]);
      t = fmaxf(t, __shfl_xor(t, 1, 32));
      t = fmaxf(t, __shfl_xor(t, 2, 32));
      t = fmaxf(t, __shfl_xor(t, 4, 32));
      t = fmaxf(t, __shfl_xor(t, 8, 32));
      float mnew = fmaxf(mrun[r], t);
      float al = __expf(mrun[r] - mnew);
      float p0 = __expf(sc[0][r] - mnew);
      float p1 = __expf(sc[1][r] - mnew);
      float rs = p0 + p1;
      rs += __shfl_xor(rs, 1, 32);
      rs += __shfl_xor(rs, 2, 32);
      rs += __shfl_xor(rs, 4, 32);
      rs += __shfl_xor(rs, 8, 32);
      lrun[r] = lrun[r] * al + rs;
      mrun[r] = mnew;
      p0h[r] = p0;
      p1h[r] = p1;
#pragma unroll
      for (int t2 = 0; t2 < 4; ++t2) oacc[t2][r] *= al;
    }

    // ---- stage P (col-major), gather A-fragment (K=32) ----
    {
      half8 pk0, pk1;
#pragma unroll
      for (int r = 0; r < 8; ++r) { pk0[r] = (_Float16)p0h[r]; pk1[r] = (_Float16)p1h[r]; }
      *(half8*)&Pst[wave][(lr)*16 + rbase] = pk0;
      *(half8*)&Pst[wave][(16 + lr) * 16 + rbase] = pk1;
    }
    v16h pa;
#pragma unroll
    for (int i = 0; i < 8; ++i) {
      pa[i]     = Pst[wave][(koff + i) * 16 + lr];
      pa[i + 8] = Pst[wave][(koff + 16 + i) * 16 + lr];
    }

    // ---- P @ V ----
#pragma unroll
    for (int t = 0; t < 4; ++t)
      oacc[t] = __builtin_amdgcn_wmma_f32_16x16x32_f16(false, pa, false, vfr[t], (short)0,
                                                       oacc[t], false, false);

    // ---- commit next tile into LDS buffers ----
#if !HAVE_ASYNC_LDS
#pragma unroll
    for (int it = 0; it < 2; ++it)
      *(half8*)&Kt[nxt][trow[it] * 64 + tch[it]] = kreg[it];
#endif
#pragma unroll
    for (int it = 0; it < 2; ++it)
#pragma unroll
      for (int e = 0; e < 8; ++e) Vt[nxt][(tch[it] + e) * 32 + trow[it]] = vreg[it][e];
#if HAVE_ASYNC_LDS
    __builtin_amdgcn_s_wait_asynccnt(0);
#endif
    __syncthreads();
  };

  for (int jt = 0; jt < NSEQ; jt += 64) {
    step(jt, 0, 1);       // compile-time buffer indices after inlining
    step(jt + 32, 1, 0);
  }

  // ---- finalize, scatter to [B*N, C] f16 ----
  int h = bh % NH;
#pragma unroll
  for (int r = 0; r < 8; ++r) {
    float inv = 1.f / lrun[r];
    int i = qbase + rbase + r;
    _Float16* op = attn_out + ((size_t)b * NSEQ + i) * CDIM + h * HD;
#pragma unroll
    for (int t = 0; t < 4; ++t) op[t * 16 + lr] = (_Float16)(oacc[t][r] * inv);
  }
}

// ---------------- output projection GEMM: 32x64 per wave, ping-pong pipelined ----------------
__global__ void __launch_bounds__(128)
k_gemm_proj(const _Float16* __restrict__ ah, const _Float16* __restrict__ wh,
            const float* __restrict__ bproj, float* __restrict__ out) {
  int lane = threadIdx.x & 31;
  int wave = threadIdx.x >> 5;
  int rowBase = (blockIdx.y * 4 + wave) * 32;
  int colBase = blockIdx.x * 64;
  int lr = lane & 15;
  int koff = (lane < 16) ? 0 : 8;
  int rbase = (lane < 16) ? 0 : 8;
  const _Float16* arow0 = ah + (size_t)(rowBase + lr) * CDIM;
  const _Float16* arow1 = ah + (size_t)(rowBase + 16 + lr) * CDIM;
  const _Float16* wrow[4];
#pragma unroll
  for (int t = 0; t < 4; ++t) wrow[t] = wh + (size_t)(colBase + t * 16 + lr) * CDIM;

  v8f acc[2][4] = {};
  v16h aF[2][2], bF[2][4];
  aF[0][0] = mk_frag(arow0 + koff, arow0 + koff + 16);
  aF[0][1] = mk_frag(arow1 + koff, arow1 + koff + 16);
#pragma unroll
  for (int t = 0; t < 4; ++t) bF[0][t] = mk_frag(wrow[t] + koff, wrow[t] + koff + 16);

  for (int k0 = 0; k0 < CDIM; k0 += 64) {
    int k1 = k0 + 32;
    aF[1][0] = mk_frag(arow0 + k1 + koff, arow0 + k1 + koff + 16);
    aF[1][1] = mk_frag(arow1 + k1 + koff, arow1 + k1 + koff + 16);
#pragma unroll
    for (int t = 0; t < 4; ++t) bF[1][t] = mk_frag(wrow[t] + k1 + koff, wrow[t] + k1 + koff + 16);
#pragma unroll
    for (int t = 0; t < 4; ++t) {
      acc[0][t] = __builtin_amdgcn_wmma_f32_16x16x32_f16(false, aF[0][0], false, bF[0][t],
                                                         (short)0, acc[0][t], false, false);
      acc[1][t] = __builtin_amdgcn_wmma_f32_16x16x32_f16(false, aF[0][1], false, bF[0][t],
                                                         (short)0, acc[1][t], false, false);
    }
    int k2 = (k0 + 64 < CDIM) ? (k0 + 64) : 0;
    aF[0][0] = mk_frag(arow0 + k2 + koff, arow0 + k2 + koff + 16);
    aF[0][1] = mk_frag(arow1 + k2 + koff, arow1 + k2 + koff + 16);
#pragma unroll
    for (int t = 0; t < 4; ++t) bF[0][t] = mk_frag(wrow[t] + k2 + koff, wrow[t] + k2 + koff + 16);
#pragma unroll
    for (int t = 0; t < 4; ++t) {
      acc[0][t] = __builtin_amdgcn_wmma_f32_16x16x32_f16(false, aF[1][0], false, bF[1][t],
                                                         (short)0, acc[0][t], false, false);
      acc[1][t] = __builtin_amdgcn_wmma_f32_16x16x32_f16(false, aF[1][1], false, bF[1][t],
                                                         (short)0, acc[1][t], false, false);
    }
  }
#pragma unroll
  for (int g = 0; g < 2; ++g)
#pragma unroll
    for (int t = 0; t < 4; ++t) {
      int c = colBase + t * 16 + lr;
      float bv = bproj[c];
#pragma unroll
      for (int r = 0; r < 8; ++r) {
        int m = rowBase + g * 16 + rbase + r;
        out[(size_t)m * CDIM + c] = acc[g][t][r] + bv;
      }
    }
}

extern "C" void kernel_launch(void* const* d_in, const int* in_sizes, int n_in,
                              void* d_out, int out_size, void* d_ws, size_t ws_size,
                              hipStream_t stream) {
  const float* x     = (const float*)d_in[0];
  const float* elev  = (const float*)d_in[1];
  const float* u     = (const float*)d_in[2];
  const float* v     = (const float*)d_in[3];
  const float* wqkv  = (const float*)d_in[4];
  const float* wproj = (const float*)d_in[5];
  const float* bproj = (const float*)d_in[6];
  const float* alpha = (const float*)d_in[7];
  const float* beta  = (const float*)d_in[8];

  char* ws = (char*)d_ws;
  _Float16* xh    = (_Float16*)(ws + 0);         // 6,291,456 B
  _Float16* wqh   = (_Float16*)(ws + 6291456);   // 3,538,944 B
  _Float16* wph   = (_Float16*)(ws + 9830400);   // 1,179,648 B
  _Float16* qb    = (_Float16*)(ws + 11010048);  // 6,291,456 B
  _Float16* kb    = (_Float16*)(ws + 17301504);  // 6,291,456 B
  _Float16* vb    = (_Float16*)(ws + 23592960);  // 6,291,456 B
  float*    wsb   = (float*)   (ws + 29884416);  //    16,384 B
  _Float16* biasb = (_Float16*)(ws + 29900800);  // 16,777,216 B
  _Float16* attnb = (_Float16*)(ws + 46678016);  // 6,291,456 B -> total ~53 MB
  float* out = (float*)d_out;

  int nx = NB * NSEQ * CDIM;
  k_f32_to_f16<<<(nx + 255) / 256, 256, 0, stream>>>(x, xh, nx);
  int nwq = C3 * CDIM;
  k_f32_to_f16<<<(nwq + 255) / 256, 256, 0, stream>>>(wqkv, wqh, nwq);
  int nwp = CDIM * CDIM;
  k_f32_to_f16<<<(nwp + 255) / 256, 256, 0, stream>>>(wproj, wph, nwp);

  k_wind<<<(NB * NSEQ + 255) / 256, 256, 0, stream>>>(u, v, wsb);

  size_t nbias = (size_t)NB * NSEQ * NSEQ;
  k_bias<<<(int)((nbias + 255) / 256), 256, 0, stream>>>(elev, wsb, alpha, beta, biasb);

  k_gemm_qkv<<<dim3(C3 / 64, (NB * NSEQ) / 128), 128, 0, stream>>>(xh, wqh, qb, kb, vb);

  k_attn<<<dim3(NSEQ / 64, NB * NH), 128, 0, stream>>>(qb, kb, vb, biasb, attnb);

  k_gemm_proj<<<dim3(CDIM / 64, (NB * NSEQ) / 128), 128, 0, stream>>>(attnb, wph, bproj, out);
}
